// DifferentialAttention_38585986187727
// MI455X (gfx1250) — matrix operand order repaired and checked by reading, not compiled, not run on hardware
//
#include <hip/hip_runtime.h>

// ---------------------------------------------------------------------------
// Types / constants
// ---------------------------------------------------------------------------
typedef __attribute__((ext_vector_type(16))) __bf16 v16bf;
typedef __attribute__((ext_vector_type(8)))  float  f32x8;
typedef __attribute__((ext_vector_type(4)))  int    v4i;

typedef __attribute__((address_space(1))) v4i* as1_v4i_ptr;
typedef __attribute__((address_space(3))) v4i* as3_v4i_ptr;

constexpr int BB   = 2;
constexpr int TT   = 2048;
constexpr int CC   = 2048;
constexpr int NHH  = 8;
constexpr int HDD  = 256;
constexpr int HHH  = 128;
constexpr int MTOT = BB * TT;                   // 4096 rows
constexpr float LAMBDA_INIT = 0.2f;
constexpr float RMS_EPS     = 1.1920929e-07f;
constexpr float QSCALE      = 0.08838834764831845f;  // 1/sqrt(128)
constexpr float LN10000     = 9.210340371976184f;

union FragB {
  v16bf v;
  uint4 q[2];
};

__device__ __forceinline__ unsigned short f2bf(float f) {
  unsigned int u = __float_as_uint(f);
  u += 0x7FFFu + ((u >> 16) & 1u);              // round-to-nearest-even
  return (unsigned short)(u >> 16);
}

__device__ __forceinline__ f32x8 zero8() {
  f32x8 v;
#pragma unroll
  for (int i = 0; i < 8; ++i) v[i] = 0.0f;
  return v;
}

__device__ __forceinline__ f32x8 wmma_bf16(const FragB& a, const FragB& b, f32x8 c) {
  return __builtin_amdgcn_wmma_f32_16x16x32_bf16(false, a.v, false, b.v,
                                                 (short)0, c, false, false);
}

// ---------------------------------------------------------------------------
// Async global->LDS copy (16 bytes per lane), CDNA5 ASYNCcnt path.
// Falls back to a VGPR round-trip if the builtin is not declared.
// ---------------------------------------------------------------------------
#if __has_builtin(__builtin_amdgcn_global_load_async_to_lds_b128)
#define HAVE_ASYNC_LDS 1
#else
#define HAVE_ASYNC_LDS 0
#endif

__device__ __forceinline__ void async_copy16(const unsigned short* __restrict__ g,
                                             unsigned short* s) {
#if HAVE_ASYNC_LDS
  __builtin_amdgcn_global_load_async_to_lds_b128(
      (as1_v4i_ptr)g, (as3_v4i_ptr)s, 0, 0);
#else
  *(uint4*)s = *(const uint4*)g;
#endif
}

__device__ __forceinline__ void wait_async0() {
#if HAVE_ASYNC_LDS
#if __has_builtin(__builtin_amdgcn_s_wait_asynccnt)
  __builtin_amdgcn_s_wait_asynccnt(0);
#else
  asm volatile("s_wait_asynccnt 0x0" ::: "memory");
#endif
#endif
}

// ---------------------------------------------------------------------------
// f32 -> bf16 conversion (vectorized, n multiple of 4)
// ---------------------------------------------------------------------------
__global__ void cvt_bf16(const float* __restrict__ src,
                         unsigned short* __restrict__ dst, int n4) {
  int i = blockIdx.x * blockDim.x + threadIdx.x;
  if (i < n4) {
    float4 v = ((const float4*)src)[i];
    ushort4 o;
    o.x = f2bf(v.x); o.y = f2bf(v.y); o.z = f2bf(v.z); o.w = f2bf(v.w);
    ((ushort4*)dst)[i] = o;
  }
}

// ---------------------------------------------------------------------------
// lambda = exp(sum lq1*lk1) - exp(sum lq2*lk2) + 0.2   (HD = 256 elements)
// ---------------------------------------------------------------------------
__global__ __launch_bounds__(256)
void lambda_kernel(const float* __restrict__ lq1, const float* __restrict__ lk1,
                   const float* __restrict__ lq2, const float* __restrict__ lk2,
                   float* __restrict__ lam) {
  __shared__ float2 red[256];
  const int i = threadIdx.x;
  red[i] = make_float2(lq1[i] * lk1[i], lq2[i] * lk2[i]);
  __syncthreads();
  for (int s = 128; s > 0; s >>= 1) {
    if (i < s) { red[i].x += red[i + s].x; red[i].y += red[i + s].y; }
    __syncthreads();
  }
  if (i == 0) lam[0] = __expf(red[0].x) - __expf(red[0].y) + LAMBDA_INIT;
}

// ---------------------------------------------------------------------------
// NT GEMM: D[M,N] = alpha * A[M,K] * B[N,K]^T, bf16 in, f32 out.
// Block 256 thr (8 waves), tile 128x128x32, wave tile 32x64 (2x4 WMMA tiles).
// Tiles are staged memory->LDS with double-buffered async-to-LDS copies:
// while tile k is consumed by WMMAs, the async engine fills tile k+1.
// ---------------------------------------------------------------------------
__global__ __launch_bounds__(256)
void gemm_nt_bf16(const unsigned short* __restrict__ A,
                  const unsigned short* __restrict__ Bm,
                  float* __restrict__ D,
                  int Mdim, int Ndim, int Kdim, float alpha) {
  __shared__ __align__(16) unsigned short sA[2][128][32];
  __shared__ __align__(16) unsigned short sB[2][128][32];
  const int tid  = threadIdx.x;
  const int lane = tid & 31;
  const int wid  = tid >> 5;
  const int wm = wid >> 1, wn = wid & 1;        // 4x2 wave grid
  const int bm = blockIdx.x * 128, bn = blockIdx.y * 128;
  const int lrow = tid >> 1;                    // 0..127
  const int lseg = (tid & 1) * 16;              // 0 or 16 halves
  const int l15 = lane & 15, lhi = lane >> 4;

  const unsigned short* gA = &A[(size_t)(bm + lrow) * Kdim + lseg];
  const unsigned short* gB = &Bm[(size_t)(bn + lrow) * Kdim + lseg];

  f32x8 acc[2][4];
#pragma unroll
  for (int mi = 0; mi < 2; ++mi)
#pragma unroll
    for (int ni = 0; ni < 4; ++ni) acc[mi][ni] = zero8();

  auto stage = [&](int kt, int buf) {
    const unsigned short* ga = gA + kt * 32;
    const unsigned short* gb = gB + kt * 32;
    async_copy16(ga,     &sA[buf][lrow][lseg]);
    async_copy16(ga + 8, &sA[buf][lrow][lseg + 8]);
    async_copy16(gb,     &sB[buf][lrow][lseg]);
    async_copy16(gb + 8, &sB[buf][lrow][lseg + 8]);
  };

  const int nk = Kdim >> 5;                     // K tiles of 32
  stage(0, 0);
  int cur = 0;

  for (int kt = 0; kt < nk; ++kt) {
    wait_async0();                              // own async writes to LDS done
    __syncthreads();                            // everyone's tile visible
    if (kt + 1 < nk) stage(kt + 1, cur ^ 1);    // prefetch next tile into LDS

    FragB a[2], b[4];
#pragma unroll
    for (int mi = 0; mi < 2; ++mi) {
      // A fragment: lane row m, halves K = koff..koff+15 contiguous (koff = lhi*16)
      const unsigned short* p = &sA[cur][wm * 32 + mi * 16 + l15][lhi * 16];
      a[mi].q[0] = *(const uint4*)p;
      a[mi].q[1] = *(const uint4*)(p + 8);
    }
#pragma unroll
    for (int ni = 0; ni < 4; ++ni) {
      // B fragment: lane col n, halves K = koff2..koff2+15 contiguous (koff2 = lhi*16)
      const unsigned short* p = &sB[cur][wn * 64 + ni * 16 + l15][lhi * 16];
      b[ni].q[0] = *(const uint4*)p;
      b[ni].q[1] = *(const uint4*)(p + 8);
    }
#pragma unroll
    for (int mi = 0; mi < 2; ++mi)
#pragma unroll
      for (int ni = 0; ni < 4; ++ni)
        acc[mi][ni] = wmma_bf16(a[mi], b[ni], acc[mi][ni]);

    __syncthreads();                            // all reads of 'cur' done
    cur ^= 1;
  }

#pragma unroll
  for (int mi = 0; mi < 2; ++mi)
#pragma unroll
    for (int ni = 0; ni < 4; ++ni) {
      const int col = bn + wn * 64 + ni * 16 + l15;
#pragma unroll
      for (int i = 0; i < 8; ++i) {
        const int row = bm + wm * 32 + mi * 16 + i + lhi * 8;
        D[(size_t)row * Ndim + col] = alpha * acc[mi][ni][i];
      }
    }
}

// ---------------------------------------------------------------------------
// RMS-norm (per 128-chunk) + per-head rotary + pack bf16.
// grid (B*T, NH), block 128. isQ folds 1/sqrt(128) into q.
// dst layout: [(b*NH+h)*T + t][128]
// ---------------------------------------------------------------------------
__global__ __launch_bounds__(128)
void pack_qk(const float* __restrict__ src,
             unsigned short* __restrict__ o1,
             unsigned short* __restrict__ o2, int isQ) {
  __shared__ float2 red2[128];
  __shared__ float sh1[128], sh2[128];
  const int bt = blockIdx.x;
  const int h  = blockIdx.y;
  const int d  = threadIdx.x;
  const int t  = bt & (TT - 1);
  const int b  = bt >> 11;                      // T = 2048

  const float* row = src + (size_t)bt * CC + h * HDD;
  const float x1 = row[d];
  const float x2 = row[HHH + d];

  red2[d] = make_float2(x1 * x1, x2 * x2);
  __syncthreads();
  for (int s = 64; s > 0; s >>= 1) {
    if (d < s) { red2[d].x += red2[d + s].x; red2[d].y += red2[d + s].y; }
    __syncthreads();
  }
  const float r1 = rsqrtf(red2[0].x * (1.0f / HHH) + RMS_EPS);
  const float r2 = rsqrtf(red2[0].y * (1.0f / HHH) + RMS_EPS);

  sh1[d] = x1 * r1;
  sh2[d] = x2 * r2;
  __syncthreads();

  // rotary: angle depends only on head index h (reference bug preserved)
  const int j = d & 63;
  const float inv_freq = __expf(-LN10000 * (float)j * (1.0f / 64.0f));
  const float ang = (float)h * inv_freq;
  const float c = __cosf(ang), sn = __sinf(ang);

  const float a1 = sh1[j], b1 = sh1[j + 64];
  const float a2 = sh2[j], b2 = sh2[j + 64];
  const float v1 = (d < 64) ? (a1 * c + b1 * sn) : (-a1 * sn + b1 * c);
  const float v2 = (d < 64) ? (a2 * c + b2 * sn) : (-a2 * sn + b2 * c);

  const float sc = isQ ? QSCALE : 1.0f;
  const size_t o = ((size_t)(b * NHH + h) * TT + t) * HHH + d;
  o1[o] = f2bf(v1 * sc);
  o2[o] = f2bf(v2 * sc);
}

// V -> bf16, transposed: vt[(b*NH+h)*HD + d][T]
__global__ __launch_bounds__(128)
void pack_v(const float* __restrict__ src, unsigned short* __restrict__ vt) {
  const int bt = blockIdx.x;
  const int h  = blockIdx.y;
  const int d  = threadIdx.x;
  const int t  = bt & (TT - 1);
  const int b  = bt >> 11;
  const float* row = src + (size_t)bt * CC + h * HDD;
  const int bh = b * NHH + h;
  vt[((size_t)bh * HDD + d) * TT + t]       = f2bf(row[d]);
  vt[((size_t)bh * HDD + d + 128) * TT + t] = f2bf(row[d + 128]);
}

// ---------------------------------------------------------------------------
// One online-softmax + PV step for one attention branch.
// s0/s1: 16x16 score tiles for keys [kb,kb+16) and [kb+16,kb+32), C layout.
// ---------------------------------------------------------------------------
__device__ __forceinline__ void online_update(
    f32x8 s0, f32x8 s1, float m[8], float l[8], f32x8 O[8],
    unsigned short (*sPw)[32], const unsigned short* __restrict__ vtp,
    int kb, int qbase, int l15, int lhi) {
  // causal mask (row = qbase + i + 8*lhi, col key = kb + tile*16 + l15)
#pragma unroll
  for (int i = 0; i < 8; ++i) {
    const int qg = qbase + i + lhi * 8;
    if (kb + l15 > qg)      s0[i] = -1e30f;
    if (kb + 16 + l15 > qg) s1[i] = -1e30f;
  }
  float rmax[8];
#pragma unroll
  for (int i = 0; i < 8; ++i) rmax[i] = fmaxf(s0[i], s1[i]);
#pragma unroll
  for (int off = 1; off < 16; off <<= 1)
#pragma unroll
    for (int i = 0; i < 8; ++i)
      rmax[i] = fmaxf(rmax[i], __shfl_xor(rmax[i], off, 32));

  float rsum[8], corr[8];
#pragma unroll
  for (int i = 0; i < 8; ++i) {
    const float nm = fmaxf(m[i], rmax[i]);
    corr[i] = __expf(m[i] - nm);
    m[i] = nm;
    const float p0 = __expf(s0[i] - nm);
    const float p1 = __expf(s1[i] - nm);
    rsum[i] = p0 + p1;
    sPw[i + lhi * 8][l15]      = f2bf(p0);
    sPw[i + lhi * 8][16 + l15] = f2bf(p1);
  }
#pragma unroll
  for (int off = 1; off < 16; off <<= 1)
#pragma unroll
    for (int i = 0; i < 8; ++i) rsum[i] += __shfl_xor(rsum[i], off, 32);
#pragma unroll
  for (int i = 0; i < 8; ++i) l[i] = l[i] * corr[i] + rsum[i];
#pragma unroll
  for (int nt = 0; nt < 8; ++nt)
#pragma unroll
    for (int i = 0; i < 8; ++i) O[nt][i] *= corr[i];

  // drain our LDS stores before re-reading in A-fragment layout (same wave)
  asm volatile("s_wait_dscnt 0" ::: "memory");

  FragB pf;
  const unsigned short* pp = &sPw[l15][lhi * 16];
  pf.q[0] = *(const uint4*)pp;
  pf.q[1] = *(const uint4*)(pp + 8);

#pragma unroll
  for (int nt = 0; nt < 8; ++nt) {
    FragB vf;
    const unsigned short* vr = vtp + (size_t)(nt * 16 + l15) * TT + kb + lhi * 16;
    vf.q[0] = *(const uint4*)vr;
    vf.q[1] = *(const uint4*)(vr + 8);
    O[nt] = wmma_bf16(pf, vf, O[nt]);
  }
}

// ---------------------------------------------------------------------------
// Differential flash attention.
// grid (B*NH, T/64, 2), block 128 (4 independent waves, 16 q-rows each,
// grid.z selects 128-dim half of the 256-dim V/output).
// ---------------------------------------------------------------------------
__global__ __launch_bounds__(128)
void flash_diff_attn(const unsigned short* __restrict__ q1,
                     const unsigned short* __restrict__ q2,
                     const unsigned short* __restrict__ k1,
                     const unsigned short* __restrict__ k2,
                     const unsigned short* __restrict__ vt,
                     const float* __restrict__ lamPtr,
                     unsigned short* __restrict__ yb) {
  __shared__ __align__(16) unsigned short sP[4][16][32];
  const int lane  = threadIdx.x & 31;
  const int w     = threadIdx.x >> 5;
  const int bh    = blockIdx.x;
  const int qbase = (blockIdx.y * 4 + w) * 16;
  const int dh    = blockIdx.z;
  const int l15 = lane & 15, lhi = lane >> 4;
  const float lam = lamPtr[0];

  const unsigned short* q1p = q1 + ((size_t)bh * TT + qbase) * HHH;
  const unsigned short* q2p = q2 + ((size_t)bh * TT + qbase) * HHH;
  const unsigned short* k1p = k1 + (size_t)bh * TT * HHH;
  const unsigned short* k2p = k2 + (size_t)bh * TT * HHH;
  const unsigned short* vtp = vt + ((size_t)bh * HDD + dh * 128) * TT;

  // resident Q fragments (4 x 32-K chunks per branch)
  FragB fq1[4], fq2[4];
  {
    const unsigned short* r1 = q1p + (size_t)l15 * HHH + lhi * 16;
    const unsigned short* r2 = q2p + (size_t)l15 * HHH + lhi * 16;
#pragma unroll
    for (int dc = 0; dc < 4; ++dc) {
      fq1[dc].q[0] = *(const uint4*)(r1 + dc * 32);
      fq1[dc].q[1] = *(const uint4*)(r1 + dc * 32 + 8);
      fq2[dc].q[0] = *(const uint4*)(r2 + dc * 32);
      fq2[dc].q[1] = *(const uint4*)(r2 + dc * 32 + 8);
    }
  }

  f32x8 O1[8], O2[8];
  float m1[8], l1[8], m2[8], l2[8];
#pragma unroll
  for (int nt = 0; nt < 8; ++nt) { O1[nt] = zero8(); O2[nt] = zero8(); }
#pragma unroll
  for (int i = 0; i < 8; ++i) { m1[i] = -1e30f; l1[i] = 0.f; m2[i] = -1e30f; l2[i] = 0.f; }

  const int kend = qbase + 16;                  // causal extent
  for (int kb = 0; kb < kend; kb += 32) {
    // hide K-stream latency for the next key block
    if (kb + 32 < kend) {
      __builtin_prefetch(k1p + (size_t)(kb + 32 + l15) * HHH, 0, 1);
      __builtin_prefetch(k2p + (size_t)(kb + 32 + l15) * HHH, 0, 1);
    }
    f32x8 sA0 = zero8(), sA1 = zero8(), sB0 = zero8(), sB1 = zero8();
#pragma unroll
    for (int dc = 0; dc < 4; ++dc) {
      FragB kf;
      const unsigned short* kr;
      kr = k1p + (size_t)(kb + l15) * HHH + dc * 32 + lhi * 16;
      kf.q[0] = *(const uint4*)kr; kf.q[1] = *(const uint4*)(kr + 8);
      sA0 = wmma_bf16(fq1[dc], kf, sA0);
      kr = k1p + (size_t)(kb + 16 + l15) * HHH + dc * 32 + lhi * 16;
      kf.q[0] = *(const uint4*)kr; kf.q[1] = *(const uint4*)(kr + 8);
      sA1 = wmma_bf16(fq1[dc], kf, sA1);
      kr = k2p + (size_t)(kb + l15) * HHH + dc * 32 + lhi * 16;
      kf.q[0] = *(const uint4*)kr; kf.q[1] = *(const uint4*)(kr + 8);
      sB0 = wmma_bf16(fq2[dc], kf, sB0);
      kr = k2p + (size_t)(kb + 16 + l15) * HHH + dc * 32 + lhi * 16;
      kf.q[0] = *(const uint4*)kr; kf.q[1] = *(const uint4*)(kr + 8);
      sB1 = wmma_bf16(fq2[dc], kf, sB1);
    }
    online_update(sA0, sA1, m1, l1, O1, sP[w], vtp, kb, qbase, l15, lhi);
    online_update(sB0, sB1, m2, l2, O2, sP[w], vtp, kb, qbase, l15, lhi);
  }

  // epilogue: y = O1/l1 - lam*O2/l2, bf16 into [B,T,C]
  const int b = bh / NHH, h = bh % NHH;
#pragma unroll
  for (int i = 0; i < 8; ++i) {
    const float inv1 = 1.0f / l1[i];
    const float inv2 = lam / l2[i];
    const int t = qbase + i + lhi * 8;
    const size_t base = ((size_t)(b * TT + t)) * CC + h * HDD + dh * 128;
#pragma unroll
    for (int nt = 0; nt < 8; ++nt) {
      const float y = O1[nt][i] * inv1 - O2[nt][i] * inv2;
      yb[base + nt * 16 + l15] = f2bf(y);
    }
  }
}

// ---------------------------------------------------------------------------
// Host orchestration
// ---------------------------------------------------------------------------
extern "C" void kernel_launch(void* const* d_in, const int* in_sizes, int n_in,
                              void* d_out, int out_size, void* d_ws, size_t ws_size,
                              hipStream_t stream) {
  (void)in_sizes; (void)n_in; (void)out_size; (void)ws_size;

  const float* x   = (const float*)d_in[0];
  const float* wq  = (const float*)d_in[1];
  const float* wk  = (const float*)d_in[2];
  const float* wv  = (const float*)d_in[3];
  const float* wo  = (const float*)d_in[4];
  const float* lq1 = (const float*)d_in[5];
  const float* lk1 = (const float*)d_in[6];
  const float* lq2 = (const float*)d_in[7];
  const float* lk2 = (const float*)d_in[8];
  float* out = (float*)d_out;

  const size_t nX  = (size_t)MTOT * CC;              // 8,388,608 elements
  const size_t nW  = (size_t)CC * CC;                // 4,194,304
  const size_t nPk = (size_t)BB * NHH * TT * HHH;    // 4,194,304
  const size_t nVt = (size_t)BB * NHH * HDD * TT;    // 8,388,608

  char* p = (char*)d_ws;
  auto alloc = [&](size_t bytes) -> char* {
    char* r = p; p += (bytes + 255) & ~(size_t)255; return r;
  };
  unsigned short* xb   = (unsigned short*)alloc(nX * 2);
  unsigned short* wqb  = (unsigned short*)alloc(nW * 2);
  unsigned short* wkb  = (unsigned short*)alloc(nW * 2);
  unsigned short* wvb  = (unsigned short*)alloc(nW * 2);
  unsigned short* wob  = (unsigned short*)alloc(nW * 2);
  float*          f32s = (float*)alloc(nX * 4);      // reused Q/K/V f32 scratch
  unsigned short* q1b  = (unsigned short*)alloc(nPk * 2);
  unsigned short* q2b  = (unsigned short*)alloc(nPk * 2);
  unsigned short* k1b  = (unsigned short*)alloc(nPk * 2);
  unsigned short* k2b  = (unsigned short*)alloc(nPk * 2);
  unsigned short* vtb  = (unsigned short*)alloc(nVt * 2);
  unsigned short* ybb  = (unsigned short*)alloc(nX * 2);
  float*          lamb = (float*)alloc(256);

  const int cvtBlk = 256;
  auto cvtGrid = [&](size_t n) { return (int)((n / 4 + cvtBlk - 1) / cvtBlk); };
  cvt_bf16<<<cvtGrid(nX), cvtBlk, 0, stream>>>(x,  xb,  (int)(nX / 4));
  cvt_bf16<<<cvtGrid(nW), cvtBlk, 0, stream>>>(wq, wqb, (int)(nW / 4));
  cvt_bf16<<<cvtGrid(nW), cvtBlk, 0, stream>>>(wk, wkb, (int)(nW / 4));
  cvt_bf16<<<cvtGrid(nW), cvtBlk, 0, stream>>>(wv, wvb, (int)(nW / 4));
  cvt_bf16<<<cvtGrid(nW), cvtBlk, 0, stream>>>(wo, wob, (int)(nW / 4));

  lambda_kernel<<<1, 256, 0, stream>>>(lq1, lk1, lq2, lk2, lamb);

  const dim3 gg(MTOT / 128, CC / 128);
  const dim3 pg(MTOT, NHH);

  // Q projection -> norm/rope pack (with 1/sqrt(128) folded into q)
  gemm_nt_bf16<<<gg, 256, 0, stream>>>(xb, wqb, f32s, MTOT, CC, CC, 1.0f);
  pack_qk<<<pg, 128, 0, stream>>>(f32s, q1b, q2b, 1);
  // K projection -> norm/rope pack
  gemm_nt_bf16<<<gg, 256, 0, stream>>>(xb, wkb, f32s, MTOT, CC, CC, 1.0f);
  pack_qk<<<pg, 128, 0, stream>>>(f32s, k1b, k2b, 0);
  // V projection -> transposed bf16 pack
  gemm_nt_bf16<<<gg, 256, 0, stream>>>(xb, wvb, f32s, MTOT, CC, CC, 1.0f);
  pack_v<<<pg, 128, 0, stream>>>(f32s, vtb);

  // differential flash attention -> y bf16 [B,T,C]
  flash_diff_attn<<<dim3(BB * NHH, TT / 64, 2), 128, 0, stream>>>(
      q1b, q2b, k1b, k2b, vtb, lamb, ybb);

  // output projection with (1 - lambda_init) folded into alpha
  gemm_nt_bf16<<<gg, 256, 0, stream>>>(ybb, wob, out, MTOT, CC, CC, 0.8f);
}
